// Encoder_48533130444814
// MI455X (gfx1250) — compile-verified, matrix operand order checked by reading, not run on hardware
//
#include <hip/hip_runtime.h>
#include <math.h>

#define N_NODES 8192
#define N_EDGES 262144
#define DIM_IN  200
#define D_HID   100
#define HEADS   2
#define KNN     25
#define A_HID   50
#define SLAB    256
#define LRELU_S 0.2f

// padded f16 operand dims
#define KP_IN   224     // round_up(200, 32)
#define KP_HID  128     // round_up(100, 32)
#define NP_OUT  256     // round_up(200, 64)  (GAT weight rows, transposed)
#define NP_HID  128     // round_up(100, 64)  (fc weight rows, transposed)

// GEMM block tiling
#define BM 64
#define BN 64
#define BK 32
#define LDK (BK + 8)    // padded LDS row (halves); rows stay 16B-aligned (80B)

typedef __attribute__((ext_vector_type(16))) _Float16 v16h;
typedef __attribute__((ext_vector_type(8)))  float    v8f;

union Frag { v16h v; uint4 u[2]; };

// ---------------------------------------------------------------------------
// NT WMMA GEMM on padded f16 operands:
//   C[M x Nn] = A16[M x Kpad] * Bt16[>=Nn x Kpad]^T          (f32 accumulate)
// Preconditions (guaranteed by the f16 staging kernels):
//   M % 64 == 0, Kpad % 32 == 0, Bt has >= round_up(Nn,64) zero-padded rows,
//   K tail zero-padded. So the whole K loop is guard-free.
// Double-buffered software pipeline: each iteration issues the NEXT tile's
// GLOBAL_LOAD_ASYNC_TO_LDS_B128 copies, then waits ASYNCcnt<=2 so those stay
// in flight across the WMMA phase (copy/compute overlap on ASYNCcnt).
// Fragments are read back as two contiguous ds_load_b128 runs per matrix
// (ISA 7.12.2 16-bit layout); the WMMA region is branch-free with full EXEC.
// ---------------------------------------------------------------------------
__global__ void __launch_bounds__(256) gemm_wmma_nt_kernel(
    const _Float16* __restrict__ A, const _Float16* __restrict__ Bt,
    float* __restrict__ C, int M, int Nn, int Kpad, int ldc,
    const float* __restrict__ bias, int relu)
{
    __shared__ _Float16 Atile[2][BM][LDK];   // [buf][m][k]
    __shared__ _Float16 Btile[2][BN][LDK];   // [buf][n][k]

    const int tid  = threadIdx.x;
    const int wave = tid >> 5;
    const int lane = tid & 31;
    const int half = lane >> 4;           // lanes 16-31 take the +8 K group
    const int r    = lane & 15;

    const int nblk = (Nn + BN - 1) / BN;
    const int row0 = (blockIdx.x / nblk) * BM;
    const int col0 = (blockIdx.x % nblk) * BN;

    const int wm = wave & 3;              // 16-row subtile (0..3)
    const int wn = wave >> 2;             // 32-col subtile (0..1)

    // async-copy assignment: thread -> (row rr, 16B chunk ch) of each tile
    const int rr = tid >> 2;              // 0..63
    const int ch = tid & 3;               // 0..3
    const _Float16* gA = A  + (size_t)(row0 + rr) * Kpad + ch * 8;
    const _Float16* gB = Bt + (size_t)(col0 + rr) * Kpad + ch * 8;
    const unsigned ldsA[2] = { (unsigned)(size_t)(&Atile[0][rr][ch * 8]),
                               (unsigned)(size_t)(&Atile[1][rr][ch * 8]) };
    const unsigned ldsB[2] = { (unsigned)(size_t)(&Btile[0][rr][ch * 8]),
                               (unsigned)(size_t)(&Btile[1][rr][ch * 8]) };

    v8f c0 = {}, c1 = {};

    // prologue: copies for tile 0 into buffer 0
    asm volatile("global_load_async_to_lds_b128 %0, %1, off"
                 :: "v"(ldsA[0]), "v"(gA) : "memory");
    asm volatile("global_load_async_to_lds_b128 %0, %1, off"
                 :: "v"(ldsB[0]), "v"(gB) : "memory");
    gA += BK; gB += BK;

    const int nk = Kpad / BK;
    int cur = 0;
    for (int ki = 0; ki < nk; ++ki) {
        if (ki + 1 < nk) {
            // prefetch next tile into the other buffer, keep it in flight
            asm volatile("global_load_async_to_lds_b128 %0, %1, off"
                         :: "v"(ldsA[cur ^ 1]), "v"(gA) : "memory");
            asm volatile("global_load_async_to_lds_b128 %0, %1, off"
                         :: "v"(ldsB[cur ^ 1]), "v"(gB) : "memory");
            gA += BK; gB += BK;
            asm volatile("s_wait_asynccnt 0x2" ::: "memory");   // current tile done
        } else {
            asm volatile("s_wait_asynccnt 0x0" ::: "memory");   // drain
        }
        __syncthreads();

        // ---- branch-free fragment build + WMMA (full EXEC) ----
        // element e<8  -> K = e     + half*8   (contiguous 8-half run)
        // element e>=8 -> K = e + 8 + half*8   (contiguous 8-half run)
        const _Float16 (*At)[LDK] = Atile[cur];
        const _Float16 (*Bl)[LDK] = Btile[cur];
        const int arow = wm * 16 + r;
        Frag fa;
        fa.u[0] = *(const uint4*)&At[arow][half * 8];
        fa.u[1] = *(const uint4*)&At[arow][16 + half * 8];
        {
            const int bcol = wn * 32 + r;
            Frag fb;
            fb.u[0] = *(const uint4*)&Bl[bcol][half * 8];
            fb.u[1] = *(const uint4*)&Bl[bcol][16 + half * 8];
            c0 = __builtin_amdgcn_wmma_f32_16x16x32_f16(false, fa.v, false, fb.v,
                                                        (short)0, c0, false, false);
        }
        {
            const int bcol = wn * 32 + 16 + r;
            Frag fb;
            fb.u[0] = *(const uint4*)&Bl[bcol][half * 8];
            fb.u[1] = *(const uint4*)&Bl[bcol][16 + half * 8];
            c1 = __builtin_amdgcn_wmma_f32_16x16x32_f16(false, fa.v, false, fb.v,
                                                        (short)0, c1, false, false);
        }
        __syncthreads();    // all waves done reading buf[cur] before it is refilled
        cur ^= 1;
    }

    // ---- epilogue: C/D layout VGPR i -> row = i + half*8, col = r ----
    const int crow0 = row0 + wm * 16;
#pragma unroll
    for (int sub = 0; sub < 2; ++sub) {
        const int col = col0 + wn * 32 + sub * 16 + r;
        if (col < Nn) {
            const float badd = bias ? bias[col] : 0.f;
#pragma unroll
            for (int i = 0; i < 8; ++i) {
                const int row = crow0 + i + half * 8;
                float v = (sub ? c1[i] : c0[i]) + badd;
                if (relu) v = fmaxf(v, 0.f);
                C[(size_t)row * ldc + col] = v;
            }
        }
    }
}

// ---------------------------------------------------------------------------
// f16 staging kernels (zero-padded)
// ---------------------------------------------------------------------------
__global__ void cvt_f16_pad_kernel(const float* __restrict__ src, _Float16* __restrict__ dst,
                                   int R, int Cc, int Cpad)
{
    int t = blockIdx.x * blockDim.x + threadIdx.x;
    if (t >= R * Cpad) return;
    const int rr = t / Cpad, cc = t % Cpad;
    dst[t] = (cc < Cc) ? (_Float16)src[(size_t)rr * Cc + cc] : (_Float16)0.f;
}

// src is [K x N] row-major; dst is [Npad x Kpad] f16 (transposed, zero-padded)
__global__ void transpose_cvt_f16_kernel(const float* __restrict__ src, _Float16* __restrict__ dst,
                                         int K, int N, int Npad, int Kpad)
{
    int t = blockIdx.x * blockDim.x + threadIdx.x;
    if (t >= Npad * Kpad) return;
    const int nn = t / Kpad, kk = t % Kpad;
    dst[t] = (nn < N && kk < K) ? (_Float16)src[(size_t)kk * N + nn] : (_Float16)0.f;
}

// ---------------------------------------------------------------------------
// Small utility kernels
// ---------------------------------------------------------------------------
__global__ void fill_kernel(float* __restrict__ p, float v, int n)
{
    int t = blockIdx.x * blockDim.x + threadIdx.x;
    if (t < n) p[t] = v;
}

__device__ __forceinline__ void atomic_max_f32(float* addr, float val)
{
    unsigned* ua = (unsigned*)addr;
    unsigned old = *ua;
    while (true) {
        float f = __uint_as_float(old);
        if (f >= val) break;
        unsigned assumed = old;
        old = atomicCAS(ua, assumed, __float_as_uint(val));
        if (old == assumed) break;
    }
}

// ---------------------------------------------------------------------------
// GAT layer pieces. xp layout: [node][head*100 + d] (stride 200).
// ---------------------------------------------------------------------------
__global__ void gat_scores_kernel(const float* __restrict__ xp,
                                  const float* __restrict__ att_src,
                                  const float* __restrict__ att_dst,
                                  float* __restrict__ a_s, float* __restrict__ a_d)
{
    int t = blockIdx.x * blockDim.x + threadIdx.x;
    if (t >= N_NODES * HEADS) return;
    const int node = t >> 1, h = t & 1;
    const float* xr = xp + (size_t)node * (HEADS * D_HID) + h * D_HID;
    float ss = 0.f, sd = 0.f;
    for (int d = 0; d < D_HID; ++d) {
        ss += xr[d] * att_src[h * D_HID + d];
        sd += xr[d] * att_dst[h * D_HID + d];
    }
    a_s[t] = ss;
    a_d[t] = sd;
}

__device__ __forceinline__ void edge_sd(const int* __restrict__ src,
                                        const int* __restrict__ dst,
                                        int e, int& s, int& d)
{
    if (e < N_EDGES) { s = src[e]; d = dst[e]; }
    else             { s = e - N_EDGES; d = s; }   // self loops
}

__global__ void gat_edge_max_kernel(const int* __restrict__ src, const int* __restrict__ dst,
                                    const float* __restrict__ a_s, const float* __restrict__ a_d,
                                    float* __restrict__ m)
{
    int e = blockIdx.x * blockDim.x + threadIdx.x;
    if (e >= N_EDGES + N_NODES) return;
    int s, d; edge_sd(src, dst, e, s, d);
#pragma unroll
    for (int h = 0; h < HEADS; ++h) {
        float v = a_s[s * HEADS + h] + a_d[d * HEADS + h];
        v = v > 0.f ? v : LRELU_S * v;
        atomic_max_f32(&m[d * HEADS + h], v);
    }
}

__global__ void gat_edge_sum_kernel(const int* __restrict__ src, const int* __restrict__ dst,
                                    const float* __restrict__ a_s, const float* __restrict__ a_d,
                                    const float* __restrict__ m, float* __restrict__ z)
{
    int e = blockIdx.x * blockDim.x + threadIdx.x;
    if (e >= N_EDGES + N_NODES) return;
    int s, d; edge_sd(src, dst, e, s, d);
#pragma unroll
    for (int h = 0; h < HEADS; ++h) {
        float v = a_s[s * HEADS + h] + a_d[d * HEADS + h];
        v = v > 0.f ? v : LRELU_S * v;
        atomicAdd(&z[d * HEADS + h], __expf(v - m[d * HEADS + h]));
    }
}

__global__ void gat_edge_acc_kernel(const int* __restrict__ src, const int* __restrict__ dst,
                                    const float* __restrict__ a_s, const float* __restrict__ a_d,
                                    const float* __restrict__ m, const float* __restrict__ z,
                                    const float* __restrict__ xp, float* __restrict__ acc)
{
    int e = blockIdx.x * blockDim.x + threadIdx.x;
    if (e >= N_EDGES + N_NODES) return;
    int s, d; edge_sd(src, dst, e, s, d);
#pragma unroll
    for (int h = 0; h < HEADS; ++h) {
        float v = a_s[s * HEADS + h] + a_d[d * HEADS + h];
        v = v > 0.f ? v : LRELU_S * v;
        const float alpha = __expf(v - m[d * HEADS + h]) / z[d * HEADS + h];
        const float* xr = xp  + (size_t)s * (HEADS * D_HID) + h * D_HID;
        float*       ar = acc + (size_t)d * (HEADS * D_HID) + h * D_HID;
        for (int c = 0; c < D_HID; ++c) atomicAdd(&ar[c], alpha * xr[c]);
    }
}

__global__ void gat_finalize_kernel(const float* __restrict__ acc, const float* __restrict__ bias,
                                    const float* __restrict__ gamma, const float* __restrict__ beta,
                                    int relu, float* __restrict__ out)
{
    int t = blockIdx.x * blockDim.x + threadIdx.x;
    if (t >= N_NODES * D_HID) return;
    const int node = t / D_HID, d = t % D_HID;
    float v = 0.5f * (acc[(size_t)node * 200 + d] + acc[(size_t)node * 200 + 100 + d]) + bias[d];
    if (gamma) v = v * (gamma[d] * rsqrtf(1.0f + 1e-5f)) + beta[d];
    if (relu)  v = fmaxf(v, 0.f);
    out[t] = v;
}

// ---------------------------------------------------------------------------
// DHG pieces
// ---------------------------------------------------------------------------
// L2-normalize rows, emitting padded f16 [node][KP_HID] directly (sim operand).
__global__ void normalize_f16_kernel(const float* __restrict__ x, _Float16* __restrict__ xnh)
{
    int node = blockIdx.x * blockDim.x + threadIdx.x;
    if (node >= N_NODES) return;
    const float* xr = x + (size_t)node * D_HID;
    float s = 0.f;
    for (int d = 0; d < D_HID; ++d) s += xr[d] * xr[d];
    const float inv = 1.0f / fmaxf(sqrtf(s), 1e-12f);
    _Float16* o = xnh + (size_t)node * KP_HID;
    for (int d = 0; d < D_HID; ++d)      o[d] = (_Float16)(xr[d] * inv);
    for (int d = D_HID; d < KP_HID; ++d) o[d] = (_Float16)0.f;
}

// Block-per-row top-25 via iterative argmax over an LDS-resident sim row.
__global__ void __launch_bounds__(256) topk_kernel(const float* __restrict__ simslab,
                                                   int* __restrict__ idx, int row0)
{
    __shared__ float row[N_NODES];
    __shared__ float rv[256];
    __shared__ int   ri[256];
    const int r = blockIdx.x;
    const float* src = simslab + (size_t)r * N_NODES;
    for (int j = threadIdx.x; j < N_NODES; j += 256) row[j] = src[j];
    __syncthreads();
    for (int k = 0; k < KNN; ++k) {
        float bv = -INFINITY; int bi = 0;
        for (int j = threadIdx.x; j < N_NODES; j += 256) {
            float v = row[j];
            if (v > bv) { bv = v; bi = j; }
        }
        rv[threadIdx.x] = bv; ri[threadIdx.x] = bi;
        __syncthreads();
        for (int s = 128; s > 0; s >>= 1) {
            if (threadIdx.x < s && rv[threadIdx.x + s] > rv[threadIdx.x]) {
                rv[threadIdx.x] = rv[threadIdx.x + s];
                ri[threadIdx.x] = ri[threadIdx.x + s];
            }
            __syncthreads();
        }
        if (threadIdx.x == 0) {
            idx[(size_t)(row0 + r) * KNN + k] = ri[0];
            row[ri[0]] = -INFINITY;
        }
        __syncthreads();
    }
}

// One block per node: region gather -> KxK mixing scores -> softmax ->
// collapsed pooling  pooled[d] = b1 + sum_j (sum_k w1[k]*mult[k][j]) * region[j][d]
// Emits padded f16 row (fc GEMM A operand) directly.
__global__ void __launch_bounds__(256) dhg_mix_kernel(
    const float* __restrict__ x, const int* __restrict__ idx,
    const float* __restrict__ Wkk, const float* __restrict__ bkk,
    const float* __restrict__ w1, const float* __restrict__ b1,
    _Float16* __restrict__ pooledh)
{
    __shared__ float region[KNN][D_HID];
    __shared__ float mult[KNN][KNN];
    __shared__ float coef[KNN];
    const int node = blockIdx.x;

    for (int t = threadIdx.x; t < KNN * D_HID; t += 256) {
        const int k = t / D_HID, d = t % D_HID;
        const int j = idx[(size_t)node * KNN + k];
        region[k][d] = x[(size_t)j * D_HID + d];
    }
    __syncthreads();

    for (int t = threadIdx.x; t < KNN * KNN; t += 256) {
        const int k = t / KNN;
        const float* wrow = Wkk + (size_t)t * D_HID;
        float s = bkk[t];
        for (int d = 0; d < D_HID; ++d) s += region[k][d] * wrow[d];
        mult[k][t % KNN] = s;
    }
    __syncthreads();

    if (threadIdx.x < KNN) {                      // row softmax
        const int k = threadIdx.x;
        float mmax = -INFINITY;
        for (int j = 0; j < KNN; ++j) mmax = fmaxf(mmax, mult[k][j]);
        float ssum = 0.f;
        for (int j = 0; j < KNN; ++j) { float e = __expf(mult[k][j] - mmax); mult[k][j] = e; ssum += e; }
        const float inv = 1.0f / ssum;
        for (int j = 0; j < KNN; ++j) mult[k][j] *= inv;
    }
    __syncthreads();

    if (threadIdx.x < KNN) {
        const int j = threadIdx.x;
        float s = 0.f;
        for (int k = 0; k < KNN; ++k) s += w1[k] * mult[k][j];
        coef[j] = s;
    }
    __syncthreads();

    if (threadIdx.x < KP_HID) {
        const int d = threadIdx.x;
        float s = 0.f;
        if (d < D_HID) {
            s = b1[0];
            for (int j = 0; j < KNN; ++j) s += coef[j] * region[j][d];
        }
        pooledh[(size_t)node * KP_HID + d] = (_Float16)s;   // padded cols stay 0
    }
}

// ---------------------------------------------------------------------------
// Attention module + final blend
// ---------------------------------------------------------------------------
__global__ void atten_kernel(const float* __restrict__ x,
                             const float* __restrict__ W1, const float* __restrict__ b1,
                             const float* __restrict__ g, const float* __restrict__ bb,
                             const float* __restrict__ W2, const float* __restrict__ b2,
                             float* __restrict__ att)
{
    int node = blockIdx.x * blockDim.x + threadIdx.x;
    if (node >= N_NODES) return;
    const float* xr = x + (size_t)node * D_HID;
    const float bns = rsqrtf(1.0f + 1e-5f);
    float acc = b2[0];
    for (int j = 0; j < A_HID; ++j) {
        float s = b1[j];
        for (int d = 0; d < D_HID; ++d) s += xr[d] * W1[d * A_HID + j];
        s = s * (g[j] * bns) + bb[j];
        const float h = 1.0f / (1.0f + __expf(-s));
        acc += h * W2[j];
    }
    att[node] = acc;
}

__global__ void combine_kernel(const float* __restrict__ feax2, const float* __restrict__ x2,
                               const float* __restrict__ att1, const float* __restrict__ att2,
                               float* __restrict__ out)
{
    int t = blockIdx.x * blockDim.x + threadIdx.x;
    if (t >= N_NODES * D_HID) return;
    const int node = t / D_HID;
    const float e1 = att1[node], e2 = att2[node];
    const float mm = fmaxf(e1, e2);
    const float w1 = __expf(e1 - mm), w2 = __expf(e2 - mm);
    const float inv = 1.0f / (w1 + w2);
    out[t] = (w1 * feax2[t] + w2 * x2[t]) * inv;
}

// ---------------------------------------------------------------------------
// Host orchestration
// ---------------------------------------------------------------------------
extern "C" void kernel_launch(void* const* d_in, const int* in_sizes, int n_in,
                              void* d_out, int out_size, void* d_ws, size_t ws_size,
                              hipStream_t stream)
{
    (void)in_sizes; (void)n_in; (void)out_size; (void)ws_size;

    const float* inputx = (const float*)d_in[0];
    const int*   eidx   = (const int*)d_in[1];
    const int*   feidx  = (const int*)d_in[2];
    const float* gW[4]  = {(const float*)d_in[4],  (const float*)d_in[8],
                           (const float*)d_in[12], (const float*)d_in[16]};
    const float* gAS[4] = {(const float*)d_in[5],  (const float*)d_in[9],
                           (const float*)d_in[13], (const float*)d_in[17]};
    const float* gAD[4] = {(const float*)d_in[6],  (const float*)d_in[10],
                           (const float*)d_in[14], (const float*)d_in[18]};
    const float* gB[4]  = {(const float*)d_in[7],  (const float*)d_in[11],
                           (const float*)d_in[15], (const float*)d_in[19]};
    const float* bn_g  = (const float*)d_in[20];
    const float* bn_b  = (const float*)d_in[21];
    const float* bn1_g = (const float*)d_in[22];
    const float* bn1_b = (const float*)d_in[23];
    const float* dP[2][6];   // Wkk, bkk, w1, b1, fcW, fcb
    for (int l = 0; l < 2; ++l)
        for (int p = 0; p < 6; ++p) dP[l][p] = (const float*)d_in[24 + l * 6 + p];
    const float* aP[2][6];   // W1, b1, g, bb, W2, b2
    for (int l = 0; l < 2; ++l)
        for (int p = 0; p < 6; ++p) aP[l][p] = (const float*)d_in[36 + l * 6 + p];

    // workspace carving (256B aligned)
    char* ws = (char*)d_ws;
    auto carve = [&](size_t bytes) { char* p = ws; ws += (bytes + 255) & ~(size_t)255; return p; };
    float*    xp      = (float*)   carve((size_t)N_NODES * 200 * 4);
    float*    a_s     = (float*)   carve((size_t)N_NODES * HEADS * 4);
    float*    a_d     = (float*)   carve((size_t)N_NODES * HEADS * 4);
    float*    mbuf    = (float*)   carve((size_t)N_NODES * HEADS * 4);
    float*    zbuf    = (float*)   carve((size_t)N_NODES * HEADS * 4);
    float*    acc     = (float*)   carve((size_t)N_NODES * 200 * 4);
    float*    feax1   = (float*)   carve((size_t)N_NODES * D_HID * 4);
    float*    x1      = (float*)   carve((size_t)N_NODES * D_HID * 4);
    float*    feax2   = (float*)   carve((size_t)N_NODES * D_HID * 4);
    float*    x2b     = (float*)   carve((size_t)N_NODES * D_HID * 4);
    int*      knnidx  = (int*)     carve((size_t)N_NODES * KNN * 4);
    float*    att1    = (float*)   carve((size_t)N_NODES * 4);
    float*    att2    = (float*)   carve((size_t)N_NODES * 4);
    float*    slab    = (float*)   carve((size_t)SLAB * N_NODES * 4);
    _Float16* inx_h   = (_Float16*)carve((size_t)N_NODES * KP_IN * 2);
    _Float16* feax1_h = (_Float16*)carve((size_t)N_NODES * KP_HID * 2);
    _Float16* x1_h    = (_Float16*)carve((size_t)N_NODES * KP_HID * 2);
    _Float16* xn_h    = (_Float16*)carve((size_t)N_NODES * KP_HID * 2);
    _Float16* pool_h  = (_Float16*)carve((size_t)N_NODES * KP_HID * 2);
    _Float16* gWt[4];
    for (int i = 0; i < 4; ++i) gWt[i] = (_Float16*)carve((size_t)NP_OUT * KP_IN * 2);
    _Float16* fcWt[2];
    for (int i = 0; i < 2; ++i) fcWt[i] = (_Float16*)carve((size_t)NP_HID * KP_HID * 2);

    auto gemm = [&](const _Float16* A, const _Float16* Bt, float* C,
                    int M, int Nn, int Kpad, int ldc, const float* bias, int relu) {
        int blocks = (M / BM) * ((Nn + BN - 1) / BN);
        gemm_wmma_nt_kernel<<<blocks, 256, 0, stream>>>(A, Bt, C, M, Nn, Kpad, ldc, bias, relu);
    };
    auto fill = [&](float* p, float v, int n) {
        fill_kernel<<<(n + 255) / 256, 256, 0, stream>>>(p, v, n);
    };
    auto cvt = [&](const float* src, _Float16* dst, int R, int Cc, int Cpad) {
        int n = R * Cpad;
        cvt_f16_pad_kernel<<<(n + 255) / 256, 256, 0, stream>>>(src, dst, R, Cc, Cpad);
    };
    auto tcvt = [&](const float* src, _Float16* dst, int K, int N, int Npad, int Kpad) {
        int n = Npad * Kpad;
        transpose_cvt_f16_kernel<<<(n + 255) / 256, 256, 0, stream>>>(src, dst, K, N, Npad, Kpad);
    };

    const int EB = (N_EDGES + N_NODES + 255) / 256;

    // one-shot f16 operand staging
    cvt(inputx, inx_h, N_NODES, DIM_IN, KP_IN);
    tcvt(gW[0], gWt[0], DIM_IN, HEADS * D_HID, NP_OUT, KP_IN);
    tcvt(gW[1], gWt[1], DIM_IN, HEADS * D_HID, NP_OUT, KP_IN);
    tcvt(gW[2], gWt[2], D_HID,  HEADS * D_HID, NP_OUT, KP_HID);
    tcvt(gW[3], gWt[3], D_HID,  HEADS * D_HID, NP_OUT, KP_HID);
    tcvt(dP[0][4], fcWt[0], D_HID, D_HID, NP_HID, KP_HID);
    tcvt(dP[1][4], fcWt[1], D_HID, D_HID, NP_HID, KP_HID);

    auto run_gat = [&](const _Float16* Xh, int kpad, const int* src, const int* dst,
                       int gi, const float* gamma, const float* beta, int relu,
                       float* dstout) {
        gemm(Xh, gWt[gi], xp, N_NODES, HEADS * D_HID, kpad, HEADS * D_HID, nullptr, 0);
        gat_scores_kernel<<<(N_NODES * HEADS + 255) / 256, 256, 0, stream>>>(xp, gAS[gi], gAD[gi], a_s, a_d);
        fill(mbuf, -INFINITY, N_NODES * HEADS);
        fill(zbuf, 0.f, N_NODES * HEADS);
        fill(acc, 0.f, N_NODES * 200);
        gat_edge_max_kernel<<<EB, 256, 0, stream>>>(src, dst, a_s, a_d, mbuf);
        gat_edge_sum_kernel<<<EB, 256, 0, stream>>>(src, dst, a_s, a_d, mbuf, zbuf);
        gat_edge_acc_kernel<<<EB, 256, 0, stream>>>(src, dst, a_s, a_d, mbuf, zbuf, xp, acc);
        gat_finalize_kernel<<<(N_NODES * D_HID + 255) / 256, 256, 0, stream>>>(acc, gB[gi], gamma, beta, relu, dstout);
    };

    auto run_dhg = [&](const float* X, int li, float* dstout) {
        normalize_f16_kernel<<<(N_NODES + 255) / 256, 256, 0, stream>>>(X, xn_h);
        for (int row0 = 0; row0 < N_NODES; row0 += SLAB) {
            gemm(xn_h + (size_t)row0 * KP_HID, xn_h, slab, SLAB, N_NODES, KP_HID,
                 N_NODES, nullptr, 0);
            topk_kernel<<<SLAB, 256, 0, stream>>>(slab, knnidx, row0);
        }
        dhg_mix_kernel<<<N_NODES, 256, 0, stream>>>(X, knnidx, dP[li][0], dP[li][1],
                                                    dP[li][2], dP[li][3], pool_h);
        gemm(pool_h, fcWt[li], dstout, N_NODES, D_HID, KP_HID, D_HID, dP[li][5], 1);
    };

    const int* fsrc = feidx;           // fea_edge_index rows
    const int* fdst = feidx + N_EDGES;
    const int* esrc = eidx;            // edge_index rows
    const int* edst = eidx + N_EDGES;

    // feax1 = relu(bn(GAT1(inputx, fea_edges)))
    run_gat(inx_h, KP_IN, fsrc, fdst, 0, bn_g, bn_b, 1, feax1);
    // x1 = relu(bn1(GAT2(inputx, edges)))
    run_gat(inx_h, KP_IN, esrc, edst, 1, bn1_g, bn1_b, 1, x1);
    // feax1 = DHG1(feax1)
    run_dhg(feax1, 0, feax1);
    cvt(feax1, feax1_h, N_NODES, D_HID, KP_HID);
    // feax2 = GAT3(feax1, fea_edges)
    run_gat(feax1_h, KP_HID, fsrc, fdst, 2, nullptr, nullptr, 0, feax2);
    // x1 = DHG2(x1)
    run_dhg(x1, 1, x1);
    cvt(x1, x1_h, N_NODES, D_HID, KP_HID);
    // x2 = GAT4(x1, edges)
    run_gat(x1_h, KP_HID, esrc, edst, 3, nullptr, nullptr, 0, x2b);
    // attention scores + softmax blend
    atten_kernel<<<(N_NODES + 255) / 256, 256, 0, stream>>>(feax2, aP[0][0], aP[0][1], aP[0][2], aP[0][3], aP[0][4], aP[0][5], att1);
    atten_kernel<<<(N_NODES + 255) / 256, 256, 0, stream>>>(x2b,   aP[1][0], aP[1][1], aP[1][2], aP[1][3], aP[1][4], aP[1][5], att2);
    combine_kernel<<<(N_NODES * D_HID + 255) / 256, 256, 0, stream>>>(feax2, x2b, att1, att2, (float*)d_out);
}
